// UGAN_34746285424720
// MI455X (gfx1250) — matrix operand order repaired and checked by reading, not compiled, not run on hardware
//
#include <hip/hip_runtime.h>
#include <hip/hip_bf16.h>

// ---------------------------------------------------------------------------
// UGAN forward for MI455X (gfx1250, wave32, WMMA).
//
//  * TemporalDecay matmul collapses to d * rowsum(W_td)  (precomputed wsum).
//  * Gpre[b,t,:] = values @ W_ih^T + b_ih + b_hh precomputed as ONE bf16 WMMA
//    GEMM (40960 x 812 x 1600); per-step rank-2 correction for the imputed
//    column (VAR) and the mask column.
//  * Per step only the recurrent GEMM h_dec @ W_hh^T (2048x400x1600) is
//    sequential; same WMMA tile kernel, epilogue fuses Gpre + corrections.
//  * Matrix math: v_wmma_f32_16x16x32_bf16 (fp32 accumulate); pointwise fp32.
//  * Tile staging: double-buffered GLOBAL_LOAD_ASYNC_TO_LDS_B128 (ASYNCcnt)
//    when the toolchain exposes the builtin; sync LDS staging fallback.
// ---------------------------------------------------------------------------

typedef __bf16 bf16;
typedef __attribute__((ext_vector_type(16))) bf16  v16bf;
typedef __attribute__((ext_vector_type(8)))  float v8f;
typedef __attribute__((ext_vector_type(4)))  int   v4i;

#define SEQ   20
#define HID   400
#define NF    812
#define VARI  811
#define BATCH 2048
#define NG    1600            // 4*HID
#define NGP   1664            // NG padded to 13*128 rows for B-tiles
#define KA    832             // NF padded to 26*32
#define KH    416             // HID padded to 13*32
#define BT    (BATCH*SEQ)     // 40960

#define BM 64
#define BN 128

#if __has_builtin(__builtin_amdgcn_global_load_async_to_lds_b128)
#define HAVE_ASYNC_LDS 1
#else
#define HAVE_ASYNC_LDS 0
#endif

#if HAVE_ASYNC_LDS
// pointer-to-int4 in global (AS1) / LDS (AS3) address spaces, per the
// builtin's prototype: (int4 AS1*, int4 AS3*, imm offset, imm cpol)
typedef __attribute__((address_space(1))) v4i* gas_v4i;
typedef __attribute__((address_space(3))) v4i* las_v4i;
#define NBUF 2
__device__ __forceinline__ void async_cp16(const bf16* g, bf16* l)
{
    __builtin_amdgcn_global_load_async_to_lds_b128((gas_v4i)g, (las_v4i)l,
                                                   0, 0);
}
__device__ __forceinline__ void wait_async0()
{
#if __has_builtin(__builtin_amdgcn_s_wait_asynccnt)
    __builtin_amdgcn_s_wait_asynccnt(0);
#else
    asm volatile("s_wait_asynccnt 0x0" ::: "memory");
#endif
}
#else
#define NBUF 1
#endif

// ---------------- bf16 WMMA GEMM:  out = A[M,K] * B[N,K]^T (+ epilogue) ----
// MODE 0: out[row,col] = acc + b_ih[col] + b_hh[col]                (Gpre)
// MODE 1: out[row,col] = acc + Gpre[(row*SEQ+t),col]
//                        + dvar[row]*W_ih[col,VAR] + m[row,t]*W_ih[col,NF]
template<int MODE>
__global__ __launch_bounds__(256)
void k_gemm(const bf16* __restrict__ A, const bf16* __restrict__ Bm,
            int lda, int kchunks,
            float* __restrict__ outp,
            const float* __restrict__ e0, const float* __restrict__ e1,
            const float* __restrict__ Gpre, const float* __restrict__ Wih,
            const float* __restrict__ dvar, const float* __restrict__ msk,
            int t)
{
    __shared__ __align__(16) bf16 As[NBUF][BM][40];   // 32 K + pad
    __shared__ __align__(16) bf16 Bs[NBUF][BN][40];

    const int tid  = threadIdx.x;
    const int lane = tid & 31;
    const int w    = tid >> 5;
    const int wm   = w & 1;           // 2 M sub-blocks of 32
    const int wn   = w >> 1;          // 4 N sub-blocks of 32
    const int r    = lane & 15;
    const int hf   = lane >> 4;
    const int m0   = blockIdx.y * BM;
    const int n0   = blockIdx.x * BN;

    // per-thread staging coordinates
    const int rowA = tid >> 2, qA = tid & 3;   // 8 bf16 (16B) per thread
    const int rowB = tid >> 1, hq = tid & 1;   // 16 bf16 (32B) per thread

    v8f acc[2][2];
    #pragma unroll
    for (int i = 0; i < 2; ++i)
        #pragma unroll
        for (int j = 0; j < 2; ++j)
            #pragma unroll
            for (int e = 0; e < 8; ++e) acc[i][j][e] = 0.f;

    // WMMA compute over staged buffer `buf`
    auto compute = [&](int buf) {
        #pragma unroll
        for (int mi = 0; mi < 2; ++mi) {
            union { v16bf v; ulonglong2 q[2]; } af;
            const int am = wm * 32 + mi * 16 + r;
            // A layout: VGPR0-3 hold K = hf*8..hf*8+7, VGPR4-7 K = 16+hf*8..
            af.q[0] = *(const ulonglong2*)&As[buf][am][hf * 8];
            af.q[1] = *(const ulonglong2*)&As[buf][am][16 + hf * 8];
            #pragma unroll
            for (int ni = 0; ni < 2; ++ni) {
                union { v16bf v; ulonglong2 q[2]; } bfv;
                const int bn = wn * 32 + ni * 16 + r;
                // B layout: lane=N, VGPR0-3 K = hf*8..hf*8+7, VGPR4-7 K = 16+hf*8..
                bfv.q[0] = *(const ulonglong2*)&Bs[buf][bn][hf * 8];
                bfv.q[1] = *(const ulonglong2*)&Bs[buf][bn][16 + hf * 8];
                acc[mi][ni] = __builtin_amdgcn_wmma_f32_16x16x32_bf16(
                    false, af.v, false, bfv.v, (short)0, acc[mi][ni],
                    false, false);
            }
        }
    };

#if HAVE_ASYNC_LDS
    // Stage chunk kc into buffer buf via direct global->LDS async copies.
    auto stage = [&](int kc, int buf) {
        const bf16* srcA = A + (long)(m0 + rowA) * lda + kc * 32 + qA * 8;
        async_cp16(srcA, &As[buf][rowA][qA * 8]);
        const bf16* srcB = Bm + (long)(n0 + rowB) * lda + kc * 32 + hq * 16;
        async_cp16(srcB,     &Bs[buf][rowB][hq * 16]);
        async_cp16(srcB + 8, &Bs[buf][rowB][hq * 16 + 8]);
    };
    stage(0, 0);
    wait_async0();
    __syncthreads();
    for (int kc = 0; kc < kchunks; ++kc) {
        const int cur = kc & 1;
        if (kc + 1 < kchunks) stage(kc + 1, cur ^ 1);  // overlap with WMMA
        compute(cur);
        wait_async0();      // our prefetch landed
        __syncthreads();    // everyone's prefetch landed / reads done
    }
#else
    for (int kc = 0; kc < kchunks; ++kc) {
        __syncthreads();
        {
            const bf16* src = A + (long)(m0 + rowA) * lda + kc * 32 + qA * 8;
            *(ulonglong2*)&As[0][rowA][qA * 8] = *(const ulonglong2*)src;
        }
        {
            const bf16* src = Bm + (long)(n0 + rowB) * lda + kc * 32 + hq * 16;
            ulonglong2* d0 = (ulonglong2*)&Bs[0][rowB][hq * 16];
            d0[0] = ((const ulonglong2*)src)[0];
            d0[1] = ((const ulonglong2*)src)[1];
        }
        __syncthreads();
        compute(0);
    }
#endif

    // Epilogue. C/D layout: element v -> M = v + 8*hf, N = r.
    #pragma unroll
    for (int mi = 0; mi < 2; ++mi) {
        #pragma unroll
        for (int ni = 0; ni < 2; ++ni) {
            const int col = n0 + wn * 32 + ni * 16 + r;
            if (col < NG) {
                if (MODE == 0) {
                    const float bias = e0[col] + e1[col];
                    #pragma unroll
                    for (int v = 0; v < 8; ++v) {
                        int row = m0 + wm * 32 + mi * 16 + hf * 8 + v;
                        outp[(long)row * NG + col] = acc[mi][ni][v] + bias;
                    }
                } else {
                    const float wv  = Wih[(long)col * (NF + 1) + VARI];
                    const float wmc = Wih[(long)col * (NF + 1) + NF];
                    #pragma unroll
                    for (int v = 0; v < 8; ++v) {
                        int row = m0 + wm * 32 + mi * 16 + hf * 8 + v; // batch
                        float g = acc[mi][ni][v]
                                + Gpre[((long)row * SEQ + t) * NG + col]
                                + dvar[row] * wv
                                + msk[(long)row * SEQ + t] * wmc;
                        outp[(long)row * NG + col] = g;
                    }
                }
            }
        }
    }
}

// ---------------- prep / pointwise kernels --------------------------------
__global__ void k_init(float* h, float* c, bf16* hdec, float* num, float* den)
{
    long i = (long)blockIdx.x * 256 + threadIdx.x;
    if (i < (long)BATCH * HID) { h[i] = 0.f; c[i] = 0.f; }
    if (i < (long)BATCH * KH)  hdec[i] = (bf16)0.f;
    if (i < SEQ)               { num[i] = 0.f; den[i] = 0.f; }
}

// f32 -> padded bf16 tile buffer (rows padded with zeros too)
__global__ void k_convert(const float* __restrict__ src, bf16* __restrict__ dst,
                          long total, int dstcols, int srcstride,
                          int validrows, int validcols)
{
    for (long i = (long)blockIdx.x * 256 + threadIdx.x; i < total;
         i += (long)gridDim.x * 256) {
        long rr = i / dstcols;
        int  k  = (int)(i % dstcols);
        float v = (rr < validrows && k < validcols)
                    ? src[rr * srcstride + k] : 0.f;
        dst[i] = (bf16)v;
    }
}

__global__ void k_wsum(const float* __restrict__ W_td, float* __restrict__ wsum)
{
    int hh = blockIdx.x * 256 + threadIdx.x;
    if (hh >= HID) return;
    float s = 0.f;
    for (int k = 0; k < NF; ++k) s += W_td[(long)hh * NF + k];
    wsum[hh] = s;
}

// combinations + missing outputs + den[t] = sum_b m
__global__ void k_prep(const float* __restrict__ deltas,
                       const float* __restrict__ masks,
                       const float* __restrict__ W1, const float* __restrict__ b1,
                       const float* __restrict__ W2, const float* __restrict__ b2,
                       float* __restrict__ out, float* __restrict__ den)
{
    long i = (long)blockIdx.x * 256 + threadIdx.x;
    if (i >= (long)BT) return;
    int t = (int)(i % SEQ);
    float d = deltas[i];
    float s = b2[0];
    #pragma unroll
    for (int j = 0; j < 5; ++j)
        s += W2[j] * fmaxf(d * W1[j] + b1[j], 0.f);
    out[1 + 2 * BT + i] = __expf(-fmaxf(s, 0.f));
    float m = masks[i];
    out[1 + 3 * BT + i] = m;
    atomicAdd(&den[t], m);
}

// Per-step: gamma decay, x_h regression dot, imputation, loss numerator.
__global__ __launch_bounds__(256)
void k_s1(const float* __restrict__ h, const float* __restrict__ deltas,
          const float* __restrict__ masks, const float* __restrict__ values,
          const float* __restrict__ wsum, const float* __restrict__ b_td,
          const float* __restrict__ W_reg, const float* __restrict__ b_reg,
          bf16* __restrict__ hdec, float* __restrict__ dvar,
          float* __restrict__ out, float* __restrict__ numt, int t)
{
    int row  = blockIdx.x * 8 + (threadIdx.x >> 5);
    int lane = threadIdx.x & 31;
    float d  = deltas[(long)row * SEQ + t];
    float dot = 0.f;
    for (int k = lane; k < HID; k += 32) {
        float g  = __expf(-fmaxf(d * wsum[k] + b_td[k], 0.f));
        float hd = h[(long)row * HID + k] * g;
        hdec[(long)row * KH + k] = (bf16)hd;
        dot += hd * W_reg[k];
    }
    #pragma unroll
    for (int off = 16; off > 0; off >>= 1) dot += __shfl_xor(dot, off, 32);
    if (lane == 0) {
        float xh = dot + b_reg[0];
        float m  = masks[(long)row * SEQ + t];
        float v  = values[((long)row * SEQ + t) * NF + VARI];
        float xv = v * m + (1.f - m) * xh;
        dvar[row] = xv - v;
        out[1 + row * SEQ + t]      = xv;   // originals (x_var)
        out[1 + BT + row * SEQ + t] = xh;   // imputations (x_h)
        atomicAdd(numt, fabsf(xv - xh) * m);
    }
}

// Per-step LSTM pointwise update.
__global__ void k_s3(const float* __restrict__ gates,
                     float* __restrict__ h, float* __restrict__ c)
{
    long i = (long)blockIdx.x * 256 + threadIdx.x;
    if (i >= (long)BATCH * HID) return;
    long b = i / HID; int j = (int)(i % HID);
    const float* g = gates + b * NG;
    float ig = 1.f / (1.f + __expf(-g[j]));
    float fg = 1.f / (1.f + __expf(-g[HID + j]));
    float gg = tanhf(g[2 * HID + j]);
    float og = 1.f / (1.f + __expf(-g[3 * HID + j]));
    float cc = fg * c[i] + ig * gg;
    c[i] = cc;
    h[i] = og * tanhf(cc);
}

__global__ void k_final(const float* __restrict__ num,
                        const float* __restrict__ den, float* __restrict__ out)
{
    if (threadIdx.x == 0 && blockIdx.x == 0) {
        float l = 0.f;
        for (int t = 0; t < SEQ; ++t) l += num[t] / (den[t] + 1e-5f);
        out[0] = l / (float)SEQ;
    }
}

// ---------------------------------------------------------------------------
extern "C" void kernel_launch(void* const* d_in, const int* in_sizes, int n_in,
                              void* d_out, int out_size, void* d_ws,
                              size_t ws_size, hipStream_t stream)
{
    const float* values = (const float*)d_in[0];
    const float* masks  = (const float*)d_in[1];
    const float* deltas = (const float*)d_in[2];
    const float* W_td   = (const float*)d_in[3];
    const float* b_td   = (const float*)d_in[4];
    const float* W_reg  = (const float*)d_in[5];
    const float* b_reg  = (const float*)d_in[6];
    const float* W1     = (const float*)d_in[7];
    const float* b1     = (const float*)d_in[8];
    const float* W2     = (const float*)d_in[9];
    const float* b2     = (const float*)d_in[10];
    const float* W_ih   = (const float*)d_in[11];
    const float* W_hh   = (const float*)d_in[12];
    const float* b_ih   = (const float*)d_in[13];
    const float* b_hh   = (const float*)d_in[14];
    float* out = (float*)d_out;
    char*  ws  = (char*)d_ws;

    // workspace layout (256B aligned)
    size_t o = 0;
    auto take = [&](size_t bytes) {
        size_t cur = o; o = (o + bytes + 255) & ~(size_t)255; return cur;
    };
    bf16*  Xbf   = (bf16*) (ws + take((size_t)BT * KA * sizeof(bf16)));
    bf16*  Wihbf = (bf16*) (ws + take((size_t)NGP * KA * sizeof(bf16)));
    bf16*  Whhbf = (bf16*) (ws + take((size_t)NGP * KH * sizeof(bf16)));
    bf16*  hdec  = (bf16*) (ws + take((size_t)BATCH * KH * sizeof(bf16)));
    float* Gpre  = (float*)(ws + take((size_t)BT * NG * sizeof(float)));
    float* gates = (float*)(ws + take((size_t)BATCH * NG * sizeof(float)));
    float* hbuf  = (float*)(ws + take((size_t)BATCH * HID * sizeof(float)));
    float* cbuf  = (float*)(ws + take((size_t)BATCH * HID * sizeof(float)));
    float* wsum  = (float*)(ws + take((size_t)HID * sizeof(float)));
    float* dvar  = (float*)(ws + take((size_t)BATCH * sizeof(float)));
    float* num   = (float*)(ws + take((size_t)SEQ * sizeof(float)));
    float* den   = (float*)(ws + take((size_t)SEQ * sizeof(float)));

    // ---- init + conversions ----
    {
        long n = (long)BATCH * KH;
        k_init<<<(int)((n + 255) / 256), 256, 0, stream>>>(hbuf, cbuf, hdec,
                                                           num, den);
    }
    {
        long tot = (long)BT * KA;
        k_convert<<<(int)((tot + 255) / 256), 256, 0, stream>>>(
            values, Xbf, tot, KA, NF, BT, NF);
    }
    {
        long tot = (long)NGP * KA;
        k_convert<<<(int)((tot + 255) / 256), 256, 0, stream>>>(
            W_ih, Wihbf, tot, KA, NF + 1, NG, NF);
    }
    {
        long tot = (long)NGP * KH;
        k_convert<<<(int)((tot + 255) / 256), 256, 0, stream>>>(
            W_hh, Whhbf, tot, KH, HID, NG, HID);
    }
    k_wsum<<<(HID + 255) / 256, 256, 0, stream>>>(W_td, wsum);
    k_prep<<<(BT + 255) / 256, 256, 0, stream>>>(deltas, masks, W1, b1, W2, b2,
                                                 out, den);

    // ---- Gpre = X @ W_ih^T + b_ih + b_hh  (one big WMMA GEMM) ----
    {
        dim3 grid((NG + BN - 1) / BN, BT / BM);   // (13, 640)
        k_gemm<0><<<grid, 256, 0, stream>>>(Xbf, Wihbf, KA, KA / 32, Gpre,
                                            b_ih, b_hh,
                                            nullptr, nullptr, nullptr, nullptr,
                                            0);
    }

    // ---- sequential timesteps ----
    for (int t = 0; t < SEQ; ++t) {
        k_s1<<<BATCH / 8, 256, 0, stream>>>(hbuf, deltas, masks, values, wsum,
                                            b_td, W_reg, b_reg, hdec, dvar,
                                            out, &num[t], t);
        dim3 grid((NG + BN - 1) / BN, BATCH / BM); // (13, 32)
        k_gemm<1><<<grid, 256, 0, stream>>>(hdec, Whhbf, KH, KH / 32, gates,
                                            nullptr, nullptr,
                                            Gpre, W_ih, dvar, masks, t);
        long n = (long)BATCH * HID;
        k_s3<<<(int)((n + 255) / 256), 256, 0, stream>>>(gates, hbuf, cbuf);
    }

    k_final<<<1, 32, 0, stream>>>(num, den, out);
}
